// MultiScaleTemporalAttention_78632261255202
// MI455X (gfx1250) — matrix operand order repaired and checked by reading, not compiled, not run on hardware
//
#include <hip/hip_runtime.h>
#include <hip/hip_bf16.h>
#include <math.h>

// ---------------------------------------------------------------------------
// Multi-scale temporal attention, restructured:
//   pre-convert x/Wqkv/Wo to bf16 once ->
//   per scale w: qkv GEMM (bf16 WMMA) -> banded scores -> window softmax+mean
//   weights -> banded weighted V combine (writes bf16 y) -> Wo GEMM ->
//   linear interp accumulate.
// GEMM double-buffers LDS and uses GLOBAL_LOAD_ASYNC_TO_LDS_B128 (ASYNCcnt
// path) when the toolchain exposes the builtin, else sync b128 + ds_store.
// ---------------------------------------------------------------------------

typedef __attribute__((ext_vector_type(16))) __bf16 v16bf;
typedef __attribute__((ext_vector_type(8)))  float  v8f;
typedef int v4i __attribute__((vector_size(16)));   // matches builtin param type
typedef unsigned short ushort_t;

#if defined(__HIP_DEVICE_COMPILE__) && defined(__has_builtin)
#  if __has_builtin(__builtin_amdgcn_global_load_async_to_lds_b128) && \
      __has_builtin(__builtin_amdgcn_s_wait_asynccnt)
#    define USE_ASYNC_LDS 1
#  endif
#endif
#ifndef USE_ASYNC_LDS
#  define USE_ASYNC_LDS 0
#endif

#define AS1 __attribute__((address_space(1)))
#define AS3 __attribute__((address_space(3)))

union FragBF { v16bf v; unsigned u[8]; };

__device__ __forceinline__ unsigned pack_bf16x2(float a, float b) {
  unsigned ua = __float_as_uint(a);
  ua = (ua + 0x7FFFu + ((ua >> 16) & 1u)) >> 16;   // RNE
  unsigned ub = __float_as_uint(b);
  ub = (ub + 0x7FFFu + ((ub >> 16) & 1u)) >> 16;
  return ua | (ub << 16);
}
__device__ __forceinline__ ushort_t f2bf(float a) {
  unsigned ua = __float_as_uint(a);
  return (ushort_t)((ua + 0x7FFFu + ((ua >> 16) & 1u)) >> 16);
}

#define BM 128
#define BN 128
#define KSTEP 32
#define LSTRU 20                 // uints per LDS row: 16 payload (32 bf16) + 4 pad
#define TILE_U (128 * LSTRU)     // uints per staged tile

// C[M,N] = A[M,K] @ W[N,K]^T + bias[N]; A,W bf16, C fp32.
__global__ __launch_bounds__(256) void wmma_gemm_bf16(
    const ushort_t* __restrict__ A, const ushort_t* __restrict__ W,
    const float* __restrict__ bias, float* __restrict__ C,
    int M, int N, int K)
{
  __shared__ __align__(16) unsigned lds[4 * TILE_U]; // buf0{A,W}, buf1{A,W}

  const int tid  = threadIdx.x;
  const int lane = tid & 31;
  const int wave = tid >> 5;
  const int lm   = lane & 15;
  const int half = lane >> 4;
  const int wm   = wave & 1;      // 2 wave rows (64 C-rows each)
  const int wn   = wave >> 1;     // 4 wave cols (32 C-cols each)

  const int m0 = blockIdx.y * BM;
  const int n0 = blockIdx.x * BN;

  unsigned* const ldsA0 = lds;
  unsigned* const ldsW0 = lds + TILE_U;

  v8f acc[4][2];
  #pragma unroll
  for (int i = 0; i < 4; ++i)
    #pragma unroll
    for (int j = 0; j < 2; ++j)
      #pragma unroll
      for (int r = 0; r < 8; ++r) acc[i][j][r] = 0.0f;

  // Staging: 128 rows x 32 bf16 per tile; thread t owns row=t>>1 and two
  // 8-element (16 B) segments sp, sp+1 of that row in both A and W tiles.
  const int srow = tid >> 1;
  const int sp   = (tid & 1) * 2;
  const int garow = (m0 + srow < M) ? (m0 + srow) : (M - 1); // clamp, rows>=M discarded
  const ushort_t* const gA = A + (size_t)garow * K;
  const ushort_t* const gW = W + (size_t)(n0 + srow) * K;

  auto stage = [&](int k0, int buf) {
    unsigned* lA = ldsA0 + buf * (2 * TILE_U) + srow * LSTRU;
    unsigned* lW = ldsW0 + buf * (2 * TILE_U) + srow * LSTRU;
    #pragma unroll
    for (int s = 0; s < 2; ++s) {
      const int seg = sp + s;
#if USE_ASYNC_LDS
      __builtin_amdgcn_global_load_async_to_lds_b128(
          (AS1 v4i*)(gA + k0 + seg * 8), (AS3 v4i*)(lA + seg * 4), 0, 0);
      __builtin_amdgcn_global_load_async_to_lds_b128(
          (AS1 v4i*)(gW + k0 + seg * 8), (AS3 v4i*)(lW + seg * 4), 0, 0);
#else
      *(uint4*)(lA + seg * 4) = *(const uint4*)(gA + k0 + seg * 8);
      *(uint4*)(lW + seg * 4) = *(const uint4*)(gW + k0 + seg * 8);
#endif
    }
  };

  const int steps = K / KSTEP;
  stage(0, 0);

  for (int it = 0; it < steps; ++it) {
    const int cur = it & 1;
    const bool pre = (it + 1) < steps;
    if (pre) stage((it + 1) * KSTEP, cur ^ 1);   // prefetch next into other buf
#if USE_ASYNC_LDS
    if (pre) __builtin_amdgcn_s_wait_asynccnt(4);   // leave the 4 just-issued in flight
    else     __builtin_amdgcn_s_wait_asynccnt(0);
#endif
    __syncthreads();   // all waves' stores to lds[cur] visible

    const unsigned* lA = ldsA0 + cur * (2 * TILE_U);
    const unsigned* lW = ldsW0 + cur * (2 * TILE_U);

    // A fragment 16x32 bf16: lane(m=lm,half); VGPR v: K=(v>>2)*16+half*8+(v&3)*2
    FragBF a[4], b[2];
    #pragma unroll
    for (int i = 0; i < 4; ++i) {
      const int row = wm * 64 + i * 16 + lm;
      #pragma unroll
      for (int v = 0; v < 8; ++v)
        a[i].u[v] = lA[row * LSTRU + ((v >> 2) * 8 + half * 4 + (v & 3))];
    }
    // B fragment 32x16 bf16: lane(n=lm,half); VGPR v: K=(v>>2)*16+half*8+(v&3)*2
    #pragma unroll
    for (int j = 0; j < 2; ++j) {
      const int col = wn * 32 + j * 16 + lm;
      #pragma unroll
      for (int v = 0; v < 8; ++v)
        b[j].u[v] = lW[col * LSTRU + ((v >> 2) * 8 + half * 4 + (v & 3))];
    }
    #pragma unroll
    for (int i = 0; i < 4; ++i)
      #pragma unroll
      for (int j = 0; j < 2; ++j)
        acc[i][j] = __builtin_amdgcn_wmma_f32_16x16x32_bf16(
            false, a[i].v, false, b[j].v, (short)0, acc[i][j], false, false);
    __syncthreads();   // done reading lds[cur] before it is restaged
  }

  // C/D layout: VGPR r -> row = half*8 + r, col = lm
  #pragma unroll
  for (int j = 0; j < 2; ++j) {
    const int col = n0 + wn * 32 + j * 16 + lm;
    const float bv = bias[col];
    #pragma unroll
    for (int i = 0; i < 4; ++i) {
      #pragma unroll
      for (int r = 0; r < 8; ++r) {
        const int row = m0 + wm * 64 + i * 16 + half * 8 + r;
        if (row < M) C[(size_t)row * N + col] = acc[i][j][r] + bv;
      }
    }
  }
}

// dst[i] = bf16(src[i]), processed as pairs (n must be even; n2 = n/2)
__global__ void f32_to_bf16_kernel(const float* __restrict__ src,
                                   ushort_t* __restrict__ dst, int n2)
{
  int i = blockIdx.x * blockDim.x + threadIdx.x;
  if (i >= n2) return;
  float2 f = ((const float2*)src)[i];
  ((unsigned*)dst)[i] = pack_bf16x2(f.x, f.y);
}

// band[b,h,t,d] = q[b,t,h,:] . k[b,t+d-(w-1),h,:] / 8    (d in [0, 2w-2])
__global__ void band_kernel(const float* __restrict__ qkv,
                            float* __restrict__ band, int w, int total)
{
  int idx = blockIdx.x * blockDim.x + threadIdx.x;
  if (idx >= total) return;
  const int bn = 2 * w - 1;
  int d = idx % bn;  int r = idx / bn;
  int t = r % 512;   r /= 512;
  int h = r % 8;     int b = r / 8;
  int t2 = t + d - (w - 1);
  float s = 0.0f;
  if (t2 >= 0 && t2 < 512) {
    const float4* q = (const float4*)(qkv + (size_t)(b * 512 + t ) * 1536 + h * 64);
    const float4* k = (const float4*)(qkv + (size_t)(b * 512 + t2) * 1536 + 512 + h * 64);
    #pragma unroll
    for (int c = 0; c < 16; ++c) {
      float4 qa = q[c], ka = k[c];
      s += qa.x * ka.x + qa.y * ka.y + qa.z * ka.z + qa.w * ka.w;
    }
    s *= 0.125f;   // 1/sqrt(hd=64)
  }
  band[((size_t)(b * 8 + h) * 512 + t) * 19 + d] = s;
}

// Per (b,l,h): softmax each of w rows from the band, average rows -> c[j]
__global__ void softmax_cw_kernel(const float* __restrict__ band,
                                  float* __restrict__ cw,
                                  int w, int L, int total)
{
  int idx = blockIdx.x * blockDim.x + threadIdx.x;
  if (idx >= total) return;                 // total = 16*L*8
  int h = idx % 8; int r = idx / 8;
  int l = r % L;   int b = r / L;
  float cacc[10];
  #pragma unroll
  for (int j = 0; j < 10; ++j) cacc[j] = 0.0f;
  const float invw = 1.0f / (float)w;
  for (int i = 0; i < w; ++i) {
    const float* brow = band + ((size_t)(b * 8 + h) * 512 + (l + i)) * 19 + (w - 1 - i);
    float e[10];
    float mx = -1e30f;
    for (int j = 0; j < w; ++j) { e[j] = brow[j]; mx = fmaxf(mx, e[j]); }
    float sum = 0.0f;
    for (int j = 0; j < w; ++j) { e[j] = __expf(e[j] - mx); sum += e[j]; }
    float sc = invw / sum;
    for (int j = 0; j < w; ++j) cacc[j] += e[j] * sc;
  }
  float* dst = cw + (size_t)idx * 10;       // idx == (b*L+l)*8 + h
  for (int j = 0; j < w; ++j) dst[j] = cacc[j];
}

// y[b,l,dcol] = sum_j c[b,l,h(dcol),j] * v[b,l+j,dcol]   -> bf16 output
__global__ void weighted_v_kernel(const float* __restrict__ qkv,
                                  const float* __restrict__ cw,
                                  ushort_t* __restrict__ y,
                                  int w, int L, int total)
{
  int idx = blockIdx.x * blockDim.x + threadIdx.x;
  if (idx >= total) return;                 // total = 16*L*512
  int dcol = idx % 512; int r = idx / 512;
  int l = r % L;        int b = r / L;
  int h = dcol >> 6;
  const float* cp = cw + ((size_t)(b * L + l) * 8 + h) * 10;
  const float* vp = qkv + (size_t)(b * 512 + l) * 1536 + 1024 + dcol;
  float acc = 0.0f;
  for (int j = 0; j < w; ++j) acc += cp[j] * vp[(size_t)j * 1536];
  y[idx] = f2bf(acc);
}

// out[b,t,d] += 0.25 * linear_interp_over_L(o)[b,t,d]
__global__ void interp_acc_kernel(const float* __restrict__ o,
                                  float* __restrict__ out, int L, int total)
{
  int idx = blockIdx.x * blockDim.x + threadIdx.x;
  if (idx >= total) return;                 // total = 16*512*512
  int dcol = idx % 512; int r = idx / 512;
  int t = r % 512;      int b = r / 512;
  float src = ((float)t + 0.5f) * ((float)L / 512.0f) - 0.5f;
  src = fminf(fmaxf(src, 0.0f), (float)(L - 1));
  int i0 = (int)src;
  int i1 = min(i0 + 1, L - 1);
  float lam = src - (float)i0;
  const float* ob = o + (size_t)b * L * 512 + dcol;
  float v = ob[(size_t)i0 * 512] * (1.0f - lam) + ob[(size_t)i1 * 512] * lam;
  out[idx] += 0.25f * v;
}

__global__ void zero_kernel(float* __restrict__ p, int n) {
  int i = blockIdx.x * blockDim.x + threadIdx.x;
  if (i < n) p[i] = 0.0f;
}

extern "C" void kernel_launch(void* const* d_in, const int* in_sizes, int n_in,
                              void* d_out, int out_size, void* d_ws, size_t ws_size,
                              hipStream_t stream)
{
  (void)in_sizes; (void)n_in; (void)ws_size;
  const int Bsz = 16, T = 512, D = 512, H = 8;
  static const int scales[4] = {2, 4, 8, 10};

  const float* x    = (const float*)d_in[0];   // [16,512,512]
  const float* Wqkv = (const float*)d_in[1];   // [4,1536,512]
  const float* bqkv = (const float*)d_in[2];   // [4,1536]
  const float* Wo   = (const float*)d_in[3];   // [4,512,512]
  const float* bo   = (const float*)d_in[4];   // [4,512]
  float* out = (float*)d_out;
  float* ws  = (float*)d_ws;

  // ---- workspace layout (all offsets 16B-aligned) ----
  const size_t QKV_E  = (size_t)Bsz * T * 3 * D;       // 12,582,912 f32
  const size_t BAND_E = (size_t)Bsz * H * T * 19;      //  1,245,184 f32
  const size_t CW_E   = (size_t)Bsz * 511 * H * 10;    //    654,080 f32
  const size_t XB_E   = (size_t)Bsz * T * D;           //  4,194,304 bf16
  const size_t WQB_E  = (size_t)4 * 3 * D * D;         //  3,145,728 bf16
  const size_t WOB_E  = (size_t)4 * D * D;             //  1,048,576 bf16

  float* qkv  = ws;
  float* band = qkv + QKV_E;
  float* cw   = band + BAND_E;
  ushort_t* xb    = (ushort_t*)(cw + CW_E);
  ushort_t* wqkvb = xb + XB_E;
  ushort_t* wob   = wqkvb + WQB_E;
  ushort_t* ybb   = wob + WOB_E;                       // [16*511, 512] bf16
  float* obuf = qkv;   // qkv is dead by output-proj time; alias it

  // ---- one-time bf16 conversions ----
  {
    int n2 = (int)(XB_E / 2);
    f32_to_bf16_kernel<<<(n2 + 255) / 256, 256, 0, stream>>>(x, xb, n2);
    n2 = (int)(WQB_E / 2);
    f32_to_bf16_kernel<<<(n2 + 255) / 256, 256, 0, stream>>>(Wqkv, wqkvb, n2);
    n2 = (int)(WOB_E / 2);
    f32_to_bf16_kernel<<<(n2 + 255) / 256, 256, 0, stream>>>(Wo, wob, n2);
  }

  zero_kernel<<<(out_size + 255) / 256, 256, 0, stream>>>(out, out_size);

  for (int s = 0; s < 4; ++s) {
    const int w = scales[s];
    const int L = T - w + 1;
    const int Mq = Bsz * T;                     // 8192

    // qkv = x @ Wqkv[s]^T + bqkv[s]
    wmma_gemm_bf16<<<dim3((3 * D) / BN, Mq / BM), 256, 0, stream>>>(
        xb, wqkvb + (size_t)s * 3 * D * D, bqkv + (size_t)s * 3 * D,
        qkv, Mq, 3 * D, D);

    const int btot = Bsz * H * T * (2 * w - 1);
    band_kernel<<<(btot + 255) / 256, 256, 0, stream>>>(qkv, band, w, btot);

    const int ctot = Bsz * L * H;
    softmax_cw_kernel<<<(ctot + 255) / 256, 256, 0, stream>>>(band, cw, w, L, ctot);

    const int ytot = Bsz * L * D;
    weighted_v_kernel<<<(ytot + 255) / 256, 256, 0, stream>>>(qkv, cw, ybb, w, L, ytot);

    // o = y @ Wo[s]^T + bo[s]
    const int Mo = Bsz * L;
    wmma_gemm_bf16<<<dim3(D / BN, (Mo + BM - 1) / BM), 256, 0, stream>>>(
        ybb, wob + (size_t)s * D * D, bo + (size_t)s * D, obuf, Mo, D, D);

    const int itot = Bsz * T * D;
    interp_acc_kernel<<<(itot + 255) / 256, 256, 0, stream>>>(obuf, out, L, itot);
  }
}
